// GCNModel_85409719648815
// MI455X (gfx1250) — matrix operand order repaired and checked by reading, not compiled, not run on hardware
//
#include <hip/hip_runtime.h>
#include <math.h>

typedef __attribute__((ext_vector_type(16))) _Float16 v16h;
typedef __attribute__((ext_vector_type(8)))  float    v8f;

#define F 128          // feature width (F_IN == HID == 128)
#define NG 64          // graphs

// ---------------------------------------------------------------------------
// degree / norm
// ---------------------------------------------------------------------------
__global__ void k_deg_init(float* deg, int n) {
    int i = blockIdx.x * blockDim.x + threadIdx.x;
    if (i < n) deg[i] = 1.0f;                        // self-loop contributes 1
}

__global__ void k_deg_accum(const int* __restrict__ dst, float* deg, int ne) {
    int e = blockIdx.x * blockDim.x + threadIdx.x;
    if (e < ne) atomicAdd(&deg[dst[e]], 1.0f);
}

__global__ void k_rsqrt(float* dinv, int n) {
    int i = blockIdx.x * blockDim.x + threadIdx.x;
    if (i < n) dinv[i] = rsqrtf(fmaxf(dinv[i], 1.0f));
}

// ---------------------------------------------------------------------------
// W (128x128 f32 row-major) -> swizzled f16 B-fragments.
// B layout for v_wmma_f32_16x16x32_f16 (32x16 B): lane L holds column
// n = L&15, hh = L>>4; elements 0..7 carry K = ks*32 + hh*8 + 0..7,
// elements 8..15 carry K = ks*32 + 16 + hh*8 + 0..7 (one v16h per lane).
// frag index: ((colTile*4 + ks)*32 + lane)*16 + e
// ---------------------------------------------------------------------------
__global__ void k_wconv(const float* __restrict__ W, _Float16* __restrict__ BF) {
    int bid = blockIdx.x;            // 0..31 : (colTile, ks)
    int ct  = bid >> 2;
    int ks  = bid & 3;
    int lane = threadIdx.x;          // 0..31
    int n  = lane & 15;
    int hh = lane >> 4;
    int k0 = ks * 32 + hh * 8;
    _Float16* out = BF + (size_t)(bid * 32 + lane) * 16;
#pragma unroll
    for (int e = 0; e < 16; ++e)
        out[e] = (_Float16)W[(size_t)(k0 + (e & 7) + ((e >> 3) << 4)) * F + ct * 16 + n];
}

// ---------------------------------------------------------------------------
// H = X @ W  via v_wmma_f32_16x16x32_f16.
// Block = 256 threads = 8 waves; block handles 16 rows, wave w handles
// column tile w (16 cols); 4 K-steps of 32 cover K = 128.
// A layout (16x32 f16): lane L -> row m = L&15, half hh = L>>4;
// elements 0..7  = X[m][ks*32 + hh*8 + 0..7]
// elements 8..15 = X[m][ks*32 + 16 + hh*8 + 0..7]
// C/D layout: lane L -> col n = L&15; vgpr v -> row (L>>4)*8 + v.
// ---------------------------------------------------------------------------
__global__ __launch_bounds__(256)
void k_gemm128(const float* __restrict__ X, const _Float16* __restrict__ BF,
               float* __restrict__ H, int nrows) {
    const int wave = threadIdx.x >> 5;       // 0..7 = column tile
    const int lane = threadIdx.x & 31;
    const int m    = lane & 15;
    const int hh   = lane >> 4;
    const long row0 = (long)blockIdx.x * 16;
    if (row0 >= nrows) return;

    const float* xr = X + (row0 + m) * F;
    v8f c = {};
#pragma unroll
    for (int ks = 0; ks < 4; ++ks) {
        const float* p = xr + ks * 32 + hh * 8;
        float4 f0 = *(const float4*)(p + 0);
        float4 f1 = *(const float4*)(p + 4);
        float4 f2 = *(const float4*)(p + 16);
        float4 f3 = *(const float4*)(p + 20);
        v16h a;
        a[0]=(_Float16)f0.x;  a[1]=(_Float16)f0.y;  a[2]=(_Float16)f0.z;  a[3]=(_Float16)f0.w;
        a[4]=(_Float16)f1.x;  a[5]=(_Float16)f1.y;  a[6]=(_Float16)f1.z;  a[7]=(_Float16)f1.w;
        a[8]=(_Float16)f2.x;  a[9]=(_Float16)f2.y;  a[10]=(_Float16)f2.z; a[11]=(_Float16)f2.w;
        a[12]=(_Float16)f3.x; a[13]=(_Float16)f3.y; a[14]=(_Float16)f3.z; a[15]=(_Float16)f3.w;

        v16h b = *(const v16h*)(BF + (size_t)((wave * 4 + ks) * 32 + lane) * 16);
        c = __builtin_amdgcn_wmma_f32_16x16x32_f16(
                /*neg_a=*/false, a, /*neg_b=*/false, b,
                /*c_mod=*/(short)0, c, /*reuse_a=*/false, /*reuse_b=*/false);
    }
    float* hr = H + (row0 + hh * 8) * F + wave * 16 + m;
#pragma unroll
    for (int v = 0; v < 8; ++v) hr[(size_t)v * F] = c[v];
}

// ---------------------------------------------------------------------------
// hs[i] = h[i] * dinv[node]   (in place on H)   and   AGG[i] = hs[i]
// (AGG seeded with the self-loop term). One thread per float4.
// ---------------------------------------------------------------------------
__global__ void k_scale_init(float4* __restrict__ H, float4* __restrict__ AGG,
                             const float* __restrict__ dinv, int nquad) {
    int i = blockIdx.x * blockDim.x + threadIdx.x;
    if (i >= nquad) return;
    int node = i >> 5;                       // 32 quads per 128-wide row
    float d = dinv[node];
    float4 h = H[i];
    h.x *= d; h.y *= d; h.z *= d; h.w *= d;
    H[i] = h;
    AGG[i] = h;
}

// ---------------------------------------------------------------------------
// Edge scatter: wave per edge; 512B coalesced gather of hs[src], 32x4
// global_atomic_add_f32 into AGG[dst].
// ---------------------------------------------------------------------------
__global__ __launch_bounds__(256)
void k_scatter(const int* __restrict__ src, const int* __restrict__ dst,
               const float* __restrict__ HS, float* __restrict__ AGG, int ne) {
    int e = blockIdx.x * 8 + (threadIdx.x >> 5);
    if (e >= ne) return;
    int lane = threadIdx.x & 31;
    int s = src[e], d = dst[e];
    float4 v = ((const float4*)(HS + (size_t)s * F))[lane];
    float* o = AGG + (size_t)d * F + lane * 4;
    atomicAdd(o + 0, v.x);
    atomicAdd(o + 1, v.y);
    atomicAdd(o + 2, v.z);
    atomicAdd(o + 3, v.w);
}

// ---------------------------------------------------------------------------
// out = elu(AGG * dinv + bias)
// ---------------------------------------------------------------------------
__global__ void k_finalize(const float4* __restrict__ AGG,
                           const float* __restrict__ dinv,
                           const float* __restrict__ bias,
                           float4* __restrict__ OUT, int nquad) {
    int i = blockIdx.x * blockDim.x + threadIdx.x;
    if (i >= nquad) return;
    int node = i >> 5;
    int fq   = (i & 31) * 4;
    float d = dinv[node];
    float4 a = AGG[i];
    float v0 = a.x * d + bias[fq + 0];
    float v1 = a.y * d + bias[fq + 1];
    float v2 = a.z * d + bias[fq + 2];
    float v3 = a.w * d + bias[fq + 3];
    float4 r;
    r.x = v0 > 0.f ? v0 : (__expf(v0) - 1.f);
    r.y = v1 > 0.f ? v1 : (__expf(v1) - 1.f);
    r.z = v2 > 0.f ? v2 : (__expf(v2) - 1.f);
    r.w = v3 > 0.f ? v3 : (__expf(v3) - 1.f);
    OUT[i] = r;
}

// ---------------------------------------------------------------------------
// pooling
// ---------------------------------------------------------------------------
__global__ void k_pool_zero(float* gsum, float* gcnt) {
    int i = blockIdx.x * blockDim.x + threadIdx.x;
    if (i < NG * F) gsum[i] = 0.0f;
    if (i < NG) gcnt[i] = 0.0f;
}

__global__ __launch_bounds__(256)
void k_pool(const float* __restrict__ X, const int* __restrict__ batch,
            float* __restrict__ gsum, float* __restrict__ gcnt, int n) {
    int node = blockIdx.x * 8 + (threadIdx.x >> 5);
    if (node >= n) return;
    int lane = threadIdx.x & 31;
    int b = batch[node];
    float4 v = ((const float4*)(X + (size_t)node * F))[lane];
    float* o = gsum + (size_t)b * F + lane * 4;
    atomicAdd(o + 0, v.x);
    atomicAdd(o + 1, v.y);
    atomicAdd(o + 2, v.z);
    atomicAdd(o + 3, v.w);
    if (lane == 0) atomicAdd(&gcnt[b], 1.0f);
}

// ---------------------------------------------------------------------------
// MLP head: g = gsum/cnt; z = relu(g@Wc1 + bc1); out = sigmoid(z@Wc2 + bc2)
// one block (64 threads) per graph.
// ---------------------------------------------------------------------------
__global__ __launch_bounds__(64)
void k_head(const float* __restrict__ gsum, const float* __restrict__ gcnt,
            const float* __restrict__ Wc1, const float* __restrict__ bc1,
            const float* __restrict__ Wc2, const float* __restrict__ bc2,
            float* __restrict__ out) {
    __shared__ float g[F];
    __shared__ float z[F / 2];
    int gr = blockIdx.x;
    int t  = threadIdx.x;                   // 0..63
    float cnt = fmaxf(gcnt[gr], 1.0f);
    g[t]      = gsum[gr * F + t] / cnt;
    g[t + 64] = gsum[gr * F + t + 64] / cnt;
    __syncthreads();
    float acc = bc1[t];
#pragma unroll 4
    for (int k = 0; k < F; ++k) acc += g[k] * Wc1[k * (F / 2) + t];
    z[t] = fmaxf(acc, 0.0f);
    __syncthreads();
    if (t == 0) {
        float s = bc2[0];
        for (int j = 0; j < F / 2; ++j) s += z[j] * Wc2[j];
        out[gr] = 1.0f / (1.0f + __expf(-s));
    }
}

// ---------------------------------------------------------------------------
// host launcher
// ---------------------------------------------------------------------------
static inline size_t align256(size_t x) { return (x + 255) & ~(size_t)255; }

extern "C" void kernel_launch(void* const* d_in, const int* in_sizes, int n_in,
                              void* d_out, int out_size, void* d_ws, size_t ws_size,
                              hipStream_t stream) {
    (void)n_in; (void)out_size; (void)ws_size;

    const float* x     = (const float*)d_in[0];
    const int*   eidx  = (const int*)d_in[1];
    const int*   batch = (const int*)d_in[2];
    const float* Ws[3] = {(const float*)d_in[3], (const float*)d_in[5], (const float*)d_in[7]};
    const float* bs[3] = {(const float*)d_in[4], (const float*)d_in[6], (const float*)d_in[8]};
    const float* Wc1 = (const float*)d_in[9];
    const float* bc1 = (const float*)d_in[10];
    const float* Wc2 = (const float*)d_in[11];
    const float* bc2 = (const float*)d_in[12];
    float* out = (float*)d_out;

    const int n  = in_sizes[0] / F;          // 100000
    const int ne = in_sizes[1] / 2;          // 1600000
    const int* esrc = eidx;
    const int* edst = eidx + ne;

    // workspace carve-up
    char* p = (char*)d_ws;
    size_t off = 0;
    float* dinv = (float*)(p + off); off = align256(off + (size_t)n * 4);
    float* B0   = (float*)(p + off); off = align256(off + (size_t)n * F * 4);
    float* B1   = (float*)(p + off); off = align256(off + (size_t)n * F * 4);
    float* AGG  = (float*)(p + off); off = align256(off + (size_t)n * F * 4);
    _Float16* BF = (_Float16*)(p + off); off = align256(off + (size_t)8 * 4 * 32 * 16 * 2);
    float* gsum = (float*)(p + off); off = align256(off + (size_t)NG * F * 4);
    float* gcnt = (float*)(p + off); off = align256(off + (size_t)NG * 4);

    const int nquad = n * (F / 4);           // float4 count of an activation buffer

    // --- degrees / symmetric norm -----------------------------------------
    k_deg_init <<<(n + 255) / 256, 256, 0, stream>>>(dinv, n);
    k_deg_accum<<<(ne + 255) / 256, 256, 0, stream>>>(edst, dinv, ne);
    k_rsqrt    <<<(n + 255) / 256, 256, 0, stream>>>(dinv, n);

    // --- 3 GCN layers (ping-pong B0/B1, activations end in B1) ------------
    const float* Xin = x;
    for (int L = 0; L < 3; ++L) {
        k_wconv  <<<32, 32, 0, stream>>>(Ws[L], BF);
        k_gemm128<<<(n + 15) / 16, 256, 0, stream>>>(Xin, BF, B0, n);
        k_scale_init<<<(nquad + 255) / 256, 256, 0, stream>>>(
            (float4*)B0, (float4*)AGG, dinv, nquad);
        k_scatter<<<(ne + 7) / 8, 256, 0, stream>>>(esrc, edst, B0, AGG, ne);
        k_finalize<<<(nquad + 255) / 256, 256, 0, stream>>>(
            (const float4*)AGG, dinv, bs[L], (float4*)B1, nquad);
        Xin = B1;
    }

    // --- global mean pool + MLP head --------------------------------------
    k_pool_zero<<<(NG * F + 255) / 256, 256, 0, stream>>>(gsum, gcnt);
    k_pool<<<(n + 7) / 8, 256, 0, stream>>>(B1, batch, gsum, gcnt, n);
    k_head<<<NG, 64, 0, stream>>>(gsum, gcnt, Wc1, bc1, Wc2, bc2, out);
}
